// WindowAttention_19619410608527
// MI455X (gfx1250) — matrix operand order repaired and checked by reading, not compiled, not run on hardware
//
#include <hip/hip_runtime.h>
#include <cstdint>

// ---------------- WMMA types & fragment helpers (CDNA5 / gfx1250, wave32) ---
typedef __attribute__((ext_vector_type(16))) _Float16 v16h;
typedef __attribute__((ext_vector_type(8)))  _Float16 v8h;
typedef __attribute__((ext_vector_type(8)))  float    v8f;

__device__ __forceinline__ v8f vzero8() { v8f z = {}; return z; }

// concat two aligned 8-half vectors -> 16-element fragment
__device__ __forceinline__ v16h ld2x8(const _Float16* p0, const _Float16* p1) {
    v8h lo = *(const v8h*)p0;
    v8h hi = *(const v8h*)p1;
    return __builtin_shufflevector(lo, hi, 0, 1, 2, 3, 4, 5, 6, 7, 8, 9, 10, 11,
                                   12, 13, 14, 15);
}
// A matrix 16x32 (MxK), row-major, ld multiple of 8 halves (16B rows).
// Element e -> K = e+16g, contiguous.
__device__ __forceinline__ v16h load_a_vec(const _Float16* a, int ld) {
    int lane = threadIdx.x & 31, g = lane >> 4, m = lane & 15;
    const _Float16* r = a + m * ld + 16 * g;
    return ld2x8(r, r + 8);
}
// B matrix 32x16 (KxN) where storage is [N][K] row-major (B = stored^T).
// Element e -> K = e+16g, contiguous.  ld multiple of 8 halves.
__device__ __forceinline__ v16h load_bt_vec(const _Float16* b, int ld) {
    int lane = threadIdx.x & 31, g = lane >> 4, n = lane & 15;
    const _Float16* r = b + n * ld + 16 * g;
    return ld2x8(r, r + 8);
}
// B matrix 32x16 (KxN), row-major [K][N] (generic, strided)
__device__ __forceinline__ v16h load_b(const _Float16* b, int ld) {
    int lane = threadIdx.x & 31, g = lane >> 4, n = lane & 15;
    v16h r;
#pragma unroll
    for (int e = 0; e < 16; ++e) r[e] = b[(e + 16 * g) * ld + n];
    return r;
}
__device__ __forceinline__ v8f wmma16(v16h a, v16h b, v8f c) {
    return __builtin_amdgcn_wmma_f32_16x16x32_f16(false, a, false, b, (short)0, c,
                                                  false, false);
}

// ---------------- TDM (Tensor Data Mover) support ---------------------------
#if defined(__gfx1250__) && __has_builtin(__builtin_amdgcn_tensor_load_to_lds) && \
    __has_builtin(__builtin_amdgcn_s_wait_tensorcnt)
#define USE_TDM 1
#else
#define USE_TDM 0
#endif

#if USE_TDM
typedef unsigned int u32x4 __attribute__((ext_vector_type(4)));
typedef int i32x4 __attribute__((ext_vector_type(4)));
typedef int i32x8 __attribute__((ext_vector_type(8)));
#define RFL(v) ((uint32_t)__builtin_amdgcn_readfirstlane((int)(v)))

__device__ __forceinline__ uint32_t lds_off(const void* p) {
    return (uint32_t)(uintptr_t)p;   // generic LDS addr low 32 bits = LDS offset
}
// 2D tile load: rows x row_dw dwords, global row stride stride_dw dwords,
// LDS rows padded by (pada+1) dwords after every 2^(padi+1) dwords.
__device__ __forceinline__ void tdm_load_2d(uint32_t ldsoff, const void* gp,
                                            uint32_t rows, uint32_t row_dw,
                                            uint32_t stride_dw, uint32_t padi,
                                            uint32_t pada) {
    uint64_t ga = (uint64_t)(uintptr_t)gp;
    u32x4 g0;
    g0[0] = RFL(1u);                                    // count=1, user mode
    g0[1] = RFL(ldsoff);                                // lds_addr (bytes)
    g0[2] = RFL((uint32_t)ga);                          // global_addr lo
    g0[3] = RFL(((uint32_t)(ga >> 32) & 0x01FFFFFFu) | 0x80000000u); // hi|type=2
    uint32_t td0 = stride_dw, td1 = rows;
    i32x8 g1;
    g1[0] = (int)RFL((2u << 16) | (1u << 20) | (padi << 22) | (pada << 25));
    g1[1] = (int)RFL((td0 & 0xFFFFu) << 16);            // tensor_dim0 lo16
    g1[2] = (int)RFL((td0 >> 16) | ((td1 & 0xFFFFu) << 16));
    g1[3] = (int)RFL((td1 >> 16) | (row_dw << 16));     // tile_dim0
    g1[4] = (int)RFL(rows);                             // tile_dim1 (tile_dim2=0)
    g1[5] = (int)RFL(stride_dw);                        // dim0_stride lo32
    g1[6] = 0;
    g1[7] = 0;
    i32x4 gz = {0, 0, 0, 0};
#if __clang_major__ >= 23
    i32x8 gz8 = {0, 0, 0, 0, 0, 0, 0, 0};
    __builtin_amdgcn_tensor_load_to_lds(g0, g1, gz, gz, gz8, 0);
#else
    __builtin_amdgcn_tensor_load_to_lds(g0, g1, gz, gz, 0);
#endif
}
#endif

// ---------------- constants ------------------------------------------------
#define DIMC   512
#define WIN_   128
#define SHIFT_ 64
#define HEADS_ 8
#define DH_    64
#define MROWS  32768   // B*T = 8*4096 rows

// ---------------- kernel: roll + LayerNorm -> f16 --------------------------
__global__ __launch_bounds__(256) void prep_xn(const float* __restrict__ x,
                                               const float* __restrict__ gamma,
                                               const float* __restrict__ beta,
                                               _Float16* __restrict__ xn) {
    int row = blockIdx.x;
    int b = row >> 12, local = row & 4095;
    int tsrc = (local + SHIFT_) & 4095;   // roll(x, -SHIFT)
    const float* xr = x + ((size_t)(b << 12) + tsrc) * DIMC;
    int t = threadIdx.x;
    float v0 = xr[t], v1 = xr[t + 256];
    float s = v0 + v1, ss = v0 * v0 + v1 * v1;
#pragma unroll
    for (int m = 1; m < 32; m <<= 1) {
        s += __shfl_xor(s, m, 32);
        ss += __shfl_xor(ss, m, 32);
    }
    __shared__ float red[16];
    if ((t & 31) == 0) { red[t >> 5] = s; red[8 + (t >> 5)] = ss; }
    __syncthreads();
    float S = 0.f, SS = 0.f;
#pragma unroll
    for (int w = 0; w < 8; ++w) { S += red[w]; SS += red[8 + w]; }
    float mu = S * (1.f / 512.f);
    float var = SS * (1.f / 512.f) - mu * mu;
    float inv = rsqrtf(var + 1e-5f);
    _Float16* o = xn + (size_t)row * DIMC;
    o[t]       = (_Float16)((v0 - mu) * inv * gamma[t] + beta[t]);
    o[t + 256] = (_Float16)((v1 - mu) * inv * gamma[t + 256] + beta[t + 256]);
}

// ------- kernel: convert+TRANSPOSE 4 weight matrices f32 -> f16 [N][K] ------
__global__ __launch_bounds__(256) void convert_w(const float* __restrict__ a,
                                                 const float* __restrict__ b,
                                                 const float* __restrict__ c,
                                                 const float* __restrict__ d,
                                                 _Float16* __restrict__ out) {
    int i = blockIdx.x * 256 + threadIdx.x;
    for (int j = i; j < 262144; j += 131072) {
        int k = j >> 9, n = j & 511;
        int tj = n * 512 + k;
        out[tj]              = (_Float16)a[j];
        out[262144 + tj]     = (_Float16)b[j];
        out[2 * 262144 + tj] = (_Float16)c[j];
        out[3 * 262144 + tj] = (_Float16)d[j];
    }
}

// ---------------- kernel: p = pos_embedding @ Wp  (255 x 512, f16) ---------
__global__ __launch_bounds__(256) void compute_p(const float* __restrict__ Wp,
                                                 _Float16* __restrict__ p16) {
    __shared__ float pe[DIMC];
    int r = blockIdx.x;                   // 0..254, rel = 127 - r
    int t = threadIdx.x;
    float rel = (float)(127 - r);
    float div = __expf((float)(2 * t) * (-9.210340371976184f / 512.f));
    float ang = rel * div;
    pe[2 * t]     = __sinf(ang);
    pe[2 * t + 1] = __cosf(ang);
    __syncthreads();
    float a0 = 0.f, a1 = 0.f;
    for (int k = 0; k < DIMC; ++k) {
        float pv = pe[k];
        a0 += pv * Wp[(size_t)k * DIMC + t];
        a1 += pv * Wp[(size_t)k * DIMC + t + 256];
    }
    p16[(size_t)r * DIMC + t]       = (_Float16)a0;
    p16[(size_t)r * DIMC + t + 256] = (_Float16)a1;
}

// ---------------- kernel: 128x128-tile WMMA GEMM, K = 512 ------------------
// A: [M][512] f16 row-major.  WtT: TRANSPOSED weights [N][K] f16.
// MODE 0: out f16 = A@W + bias.  MODE 1: out f32, rows rolled (+SHIFT).
#define GM_LD 40   // LDS row stride (halves): 16 dwords data + 4 dwords pad
template <int MODE>
__global__ __launch_bounds__(256) void gemm512(const _Float16* __restrict__ A,
                                               const _Float16* __restrict__ WtT,
                                               const float* __restrict__ bias,
                                               void* __restrict__ Out) {
    __shared__ __align__(16) _Float16 sA[2][128 * GM_LD];
    __shared__ __align__(16) _Float16 sBT[2][128 * GM_LD];
    int bm = blockIdx.x, bn = blockIdx.y;
    int t = threadIdx.x;
    int wave = t >> 5, wm = wave >> 2, wn = wave & 3;   // 2x4 wave grid
    v8f acc[4][2];
#pragma unroll
    for (int i = 0; i < 4; ++i)
#pragma unroll
        for (int j = 0; j < 2; ++j) acc[i][j] = vzero8();

    const _Float16* Abase = A + (size_t)bm * 128 * DIMC;
    const _Float16* Bbase = WtT + (size_t)bn * 128 * DIMC;

    auto compute_tile = [&](const _Float16* tA, const _Float16* tB) {
#pragma unroll
        for (int i = 0; i < 4; ++i) {
            v16h a = load_a_vec(&tA[(wm * 64 + i * 16) * GM_LD], GM_LD);
#pragma unroll
            for (int j = 0; j < 2; ++j) {
                v16h b = load_bt_vec(&tB[(wn * 32 + j * 16) * GM_LD], GM_LD);
                acc[i][j] = wmma16(a, b, acc[i][j]);
            }
        }
    };

#if USE_TDM
    // double-buffered: TDM for tile it+1 overlaps WMMA on tile it
    if (wave == 0) {
        tdm_load_2d(lds_off(&sA[0][0]), Abase, 128, 16, 256, 3, 3);
        tdm_load_2d(lds_off(&sBT[0][0]), Bbase, 128, 16, 256, 3, 3);
    }
    for (int it = 0; it < 16; ++it) {
        if (wave == 0) {
            if (it < 15) {
                int nb = (it + 1) & 1, k0n = (it + 1) * 32;
                tdm_load_2d(lds_off(&sA[nb][0]), Abase + k0n, 128, 16, 256, 3, 3);
                tdm_load_2d(lds_off(&sBT[nb][0]), Bbase + k0n, 128, 16, 256, 3, 3);
                __builtin_amdgcn_s_wait_tensorcnt(2);   // tile `it` complete
            } else {
                __builtin_amdgcn_s_wait_tensorcnt(0);
            }
        }
        __syncthreads();
        compute_tile(sA[it & 1], sBT[it & 1]);
        __syncthreads();
    }
#else
    for (int k0 = 0; k0 < DIMC; k0 += 32) {
        __syncthreads();
        {   // A tile: 128 rows x 32 halves
            int row = t >> 1, co = (t & 1) * 16;
            const uint32_t* src = (const uint32_t*)(Abase + (size_t)row * DIMC + k0 + co);
            uint32_t* dst = (uint32_t*)&sA[0][row * GM_LD + co];
#pragma unroll
            for (int i = 0; i < 8; ++i) dst[i] = src[i];
        }
        {   // BT tile: 128 rows x 32 halves
            int row = t >> 1, co = (t & 1) * 16;
            const uint32_t* src = (const uint32_t*)(Bbase + (size_t)row * DIMC + k0 + co);
            uint32_t* dst = (uint32_t*)&sBT[0][row * GM_LD + co];
#pragma unroll
            for (int i = 0; i < 8; ++i) dst[i] = src[i];
        }
        __syncthreads();
        compute_tile(sA[0], sBT[0]);
    }
#endif

    int lane = t & 31, g = lane >> 4, n = lane & 15;
#pragma unroll
    for (int i = 0; i < 4; ++i)
#pragma unroll
        for (int j = 0; j < 2; ++j)
#pragma unroll
            for (int r = 0; r < 8; ++r) {
                int row = bm * 128 + wm * 64 + i * 16 + r + 8 * g;
                int col = bn * 128 + wn * 32 + j * 16 + n;
                float vv = acc[i][j][r] + bias[col];
                if (MODE == 0) {
                    ((_Float16*)Out)[(size_t)row * DIMC + col] = (_Float16)vv;
                } else {
                    int b = row >> 12, tp = row & 4095;
                    int td = (tp + SHIFT_) & 4095;   // roll(out, +SHIFT)
                    ((float*)Out)[((size_t)(b << 12) + td) * DIMC + col] = vv;
                }
            }
}

// ---------------- kernel: fused window attention ----------------------------
#define KLD 72     // k/v LDS stride (halves): 32 dwords + 4 pad  (16B rows)
#define PLD 72
#define ALD 136    // attn stride: 16B-aligned, conflict-free (68 dwords)
__global__ __launch_bounds__(256) void attention(
    const _Float16* __restrict__ qbuf, const _Float16* __restrict__ kbuf,
    const _Float16* __restrict__ vbuf, const _Float16* __restrict__ p16,
    const float* __restrict__ ubias, const float* __restrict__ vbias,
    _Float16* __restrict__ ctx) {
    __shared__ __align__(16) _Float16 sK[128 * KLD];   // later reused for V
    __shared__ __align__(16) _Float16 sP[256 * PLD];   // later reused for attn
    int nwin = blockIdx.x, h = blockIdx.y;
    int t = threadIdx.x;
    int wave = t >> 5, lane = t & 31, g = lane >> 4, nl = lane & 15;
    int qbase = wave * 16;
    size_t rowbase = (size_t)nwin * WIN_;

#if USE_TDM
    if (wave == 0) {
        tdm_load_2d(lds_off(sK), kbuf + rowbase * DIMC + h * DH_, 128, 32, 256, 4, 3);
        tdm_load_2d(lds_off(sP), p16 + h * DH_, 255, 32, 256, 4, 3);
    }
    if (t < 32) ((uint32_t*)&sP[255 * PLD])[t] = 0u;   // zero pad row 255
#else
    {   // K tile
        int row = t >> 1, off = (t & 1) * 32;
        const uint32_t* src = (const uint32_t*)(kbuf + (rowbase + row) * DIMC + h * DH_ + off);
        uint32_t* dst = (uint32_t*)&sK[row * KLD + off];
#pragma unroll
        for (int i = 0; i < 16; ++i) dst[i] = src[i];
    }
    {   // P tile + zero row 255
        int row = t;
        uint32_t* dst = (uint32_t*)&sP[row * PLD];
        if (row < 255) {
            const uint32_t* src = (const uint32_t*)(p16 + (size_t)row * DIMC + h * DH_);
#pragma unroll
            for (int i = 0; i < 32; ++i) dst[i] = src[i];
        } else {
#pragma unroll
            for (int i = 0; i < 32; ++i) dst[i] = 0u;
        }
    }
#endif

    // Q fragments (vectorized from global) + u/v bias folded in
    const _Float16* qrow = qbuf + (rowbase + qbase + nl) * DIMC + h * DH_;
    v16h q0 = ld2x8(qrow + 16 * g, qrow + 8 + 16 * g);
    v16h q1 = ld2x8(qrow + 32 + 16 * g, qrow + 40 + 16 * g);
    v16h ub0, ub1, vb0, vb1;
#pragma unroll
    for (int e = 0; e < 8; ++e) {
        int base = h * DH_ + 16 * g + e;
        ub0[e] = (_Float16)ubias[base];      ub0[e + 8] = (_Float16)ubias[base + 8];
        ub1[e] = (_Float16)ubias[base + 32]; ub1[e + 8] = (_Float16)ubias[base + 40];
        vb0[e] = (_Float16)vbias[base];      vb0[e + 8] = (_Float16)vbias[base + 8];
        vb1[e] = (_Float16)vbias[base + 32]; vb1[e + 8] = (_Float16)vbias[base + 40];
    }
    v16h au0 = q0 + ub0, au1 = q1 + ub1, av0 = q0 + vb0, av1 = q1 + vb1;

#if USE_TDM
    if (wave == 0) __builtin_amdgcn_s_wait_tensorcnt(0);
#endif
    __syncthreads();

    auto pos_tile = [&](int ct) -> v8f {
        v8f acc = vzero8();
        acc = wmma16(av0, load_bt_vec(&sP[(ct * 16) * PLD], PLD), acc);
        acc = wmma16(av1, load_bt_vec(&sP[(ct * 16) * PLD + 32], PLD), acc);
        return acc;
    };

    const float scale = 0.04419417382415922f;   // 1/sqrt(512)
    v8f s[8];
    v8f posA = pos_tile(7 - wave);
#pragma unroll
    for (int kt = 0; kt < 8; ++kt) {
        v8f posB = pos_tile(8 - wave + kt);
        v8f c = vzero8();
        c = wmma16(au0, load_bt_vec(&sK[(kt * 16) * KLD], KLD), c);
        c = wmma16(au1, load_bt_vec(&sK[(kt * 16) * KLD + 32], KLD), c);
#pragma unroll
        for (int r = 0; r < 8; ++r) {
            int rl = nl + 15 - r - 8 * g;            // pos gather: r_idx=k-q+127
            int src = (g << 4) | (rl & 15);
            float pa = __shfl(posA[r], src, 32);
            float pb = __shfl(posB[r], src, 32);
            float pos = (rl < 16) ? pa : pb;
            int qg = qbase + r + 8 * g, kg = kt * 16 + nl;
            float sc = (c[r] + pos) * scale;
            if (qg != kg) sc -= 100.0f;              // degenerate arange-id mask
            s[kt][r] = sc;
        }
        posA = posB;
    }

    // softmax (each row lives within one 16-lane half-wave)
#pragma unroll
    for (int r = 0; r < 8; ++r) {
        float mx = s[0][r];
#pragma unroll
        for (int kt = 1; kt < 8; ++kt) mx = fmaxf(mx, s[kt][r]);
#pragma unroll
        for (int m = 1; m < 16; m <<= 1) mx = fmaxf(mx, __shfl_xor(mx, m, 32));
        float sum = 0.f;
#pragma unroll
        for (int kt = 0; kt < 8; ++kt) {
            float e = __expf(s[kt][r] - mx);
            s[kt][r] = e;
            sum += e;
        }
#pragma unroll
        for (int m = 1; m < 16; m <<= 1) sum += __shfl_xor(sum, m, 32);
        float inv = 1.f / sum;
#pragma unroll
        for (int kt = 0; kt < 8; ++kt) s[kt][r] *= inv;
    }
    __syncthreads();   // all waves done reading sK / sP

    // V overlays sK; attn (f16, ld ALD) overlays sP
    _Float16* sV = sK;
    _Float16* sAt = sP;
#if USE_TDM
    if (wave == 0)
        tdm_load_2d(lds_off(sV), vbuf + rowbase * DIMC + h * DH_, 128, 32, 256, 4, 3);
#else
    {
        int row = t >> 1, off = (t & 1) * 32;
        const uint32_t* src = (const uint32_t*)(vbuf + (rowbase + row) * DIMC + h * DH_ + off);
        uint32_t* dst = (uint32_t*)&sV[row * KLD + off];
#pragma unroll
        for (int i = 0; i < 16; ++i) dst[i] = src[i];
    }
#endif
#pragma unroll
    for (int kt = 0; kt < 8; ++kt)
#pragma unroll
        for (int r = 0; r < 8; ++r)
            sAt[(qbase + r + 8 * g) * ALD + kt * 16 + nl] = (_Float16)s[kt][r];
#if USE_TDM
    if (wave == 0) __builtin_amdgcn_s_wait_tensorcnt(0);
#endif
    __syncthreads();

    // ctx = attn @ V : per wave [16][64]
    v8f acc[4];
#pragma unroll
    for (int dt = 0; dt < 4; ++dt) acc[dt] = vzero8();
#pragma unroll
    for (int kc = 0; kc < 4; ++kc) {
        v16h a = load_a_vec(&sAt[qbase * ALD + kc * 32], ALD);
#pragma unroll
        for (int dt = 0; dt < 4; ++dt) {
            v16h b = load_b(&sV[(kc * 32) * KLD + dt * 16], KLD);
            acc[dt] = wmma16(a, b, acc[dt]);
        }
    }
#pragma unroll
    for (int dt = 0; dt < 4; ++dt)
#pragma unroll
        for (int r = 0; r < 8; ++r)
            ctx[(rowbase + qbase + r + 8 * g) * DIMC + h * DH_ + dt * 16 + nl] =
                (_Float16)acc[dt][r];
}

// ---------------- host launch ----------------------------------------------
extern "C" void kernel_launch(void* const* d_in, const int* in_sizes, int n_in,
                              void* d_out, int out_size, void* d_ws, size_t ws_size,
                              hipStream_t stream) {
    const float* x     = (const float*)d_in[0];
    const float* gamma = (const float*)d_in[1];
    const float* beta  = (const float*)d_in[2];
    const float* Wq = (const float*)d_in[3];
    const float* bq = (const float*)d_in[4];
    const float* Wk = (const float*)d_in[5];
    const float* bk = (const float*)d_in[6];
    const float* Wv = (const float*)d_in[7];
    const float* bv = (const float*)d_in[8];
    const float* Wp = (const float*)d_in[9];
    const float* Wo = (const float*)d_in[10];
    const float* bo = (const float*)d_in[11];
    const float* ub = (const float*)d_in[12];
    const float* vb = (const float*)d_in[13];

    char* ws = (char*)d_ws;
    const size_t MAT = (size_t)MROWS * DIMC * sizeof(_Float16);   // 32 MB
    _Float16* xn   = (_Float16*)(ws);
    _Float16* qbuf = (_Float16*)(ws + MAT);
    _Float16* kbuf = (_Float16*)(ws + 2 * MAT);
    _Float16* vbuf = (_Float16*)(ws + 3 * MAT);
    _Float16* ctx  = (_Float16*)(ws + 4 * MAT);
    _Float16* w16  = (_Float16*)(ws + 5 * MAT);   // 4 transposed matrices
    _Float16* wq16 = w16;
    _Float16* wk16 = w16 + 262144;
    _Float16* wv16 = w16 + 2 * 262144;
    _Float16* wo16 = w16 + 3 * 262144;
    _Float16* p16  = (_Float16*)(ws + 5 * MAT + 4 * 262144 * sizeof(_Float16));

    convert_w<<<dim3(512), 256, 0, stream>>>(Wq, Wk, Wv, Wo, w16);
    prep_xn<<<dim3(MROWS), 256, 0, stream>>>(x, gamma, beta, xn);
    compute_p<<<dim3(255), 256, 0, stream>>>(Wp, p16);

    dim3 gg(MROWS / 128, DIMC / 128);   // (256, 4)
    gemm512<0><<<gg, 256, 0, stream>>>(xn, wq16, bq, qbuf);
    gemm512<0><<<gg, 256, 0, stream>>>(xn, wk16, bk, kbuf);
    gemm512<0><<<gg, 256, 0, stream>>>(xn, wv16, bv, vbuf);

    attention<<<dim3(256, HEADS_), 256, 0, stream>>>(qbuf, kbuf, vbuf, p16, ub, vb, ctx);

    gemm512<1><<<gg, 256, 0, stream>>>(ctx, wo16, bo, d_out);
}
